// GATv1_7395933684242
// MI455X (gfx1250) — matrix operand order repaired and checked by reading, not compiled, not run on hardware
//
#include <hip/hip_runtime.h>
#include <hip/hip_bf16.h>

// ---------------------------------------------------------------------------
// GAT(128->3x64) -> ReLU -> GAT(192->3x64) -> ReLU -> SAGE(192->192) ->
// MLP(192->64->16) -> sigmoid, on gfx1250 (wave32, WMMA 16x16x32 f16).
// Dense projections use v_wmma_f32_16x16x32_f16 with f32 accumulation,
// fragments loaded directly from global (everything lives in the 192MB L2).
// Edge softmax/aggregation via float atomics (order-preserving-uint max).
// GEMM processes column tiles in groups of 4 to stay spill-free (~60 VGPRs).
// ---------------------------------------------------------------------------

#define NNODES 50000
#define NEDGES 800000
#define NEDGES_SL (NEDGES + NNODES)   // with self-loops
#define FEAT 192                      // HEADS*HID
#define INC 128
#define HID 64
#define HEADS 3
#define OUTC 16
#define NEG_SLOPE 0.2f

typedef __attribute__((ext_vector_type(16))) _Float16 v16h;
typedef __attribute__((ext_vector_type(8)))  _Float16 v8h;
typedef __attribute__((ext_vector_type(8)))  float    v8f;

static inline int cdiv(int a, int b) { return (a + b - 1) / b; }

// ---------------- order-preserving float<->uint for atomic max ------------
__device__ __forceinline__ unsigned fenc(float f) {
    unsigned u = __float_as_uint(f);
    return (u & 0x80000000u) ? ~u : (u | 0x80000000u);
}
__device__ __forceinline__ float fdec(unsigned u) {
    u = (u & 0x80000000u) ? (u & 0x7FFFFFFFu) : ~u;
    return __uint_as_float(u);
}
#define FENC_NEG_INF 0x007FFFFFu   // fenc(-inf)

// ---------------------------------------------------------------------------
// WMMA GEMM: C[M,N] = act( A_f16[M,K] @ B (given as Bt_f16[N,K]) + bias
//                          (+ C_old if accum) )
// Each wave computes a 16-row strip across all NT=N/16 column tiles,
// processed GROUP tiles at a time so accumulators never spill.
// M % 16 == 0, K % 32 == 0 for this workload (M=50000, K in {128,192,64}).
// ---------------------------------------------------------------------------
template <int NT>
__global__ __launch_bounds__(256, 1)
void gemm_wmma_kernel(const _Float16* __restrict__ A,
                      const _Float16* __restrict__ Bt,
                      float* __restrict__ C,
                      const float* __restrict__ bias,
                      int M, int K, int Ncols, int accum, int act) {
    constexpr int GROUP = (NT >= 4) ? 4 : NT;   // col tiles in flight
    const int wave = blockIdx.x * (blockDim.x >> 5) + (threadIdx.x >> 5);
    const int lane = threadIdx.x & 31;
    const int m0 = wave * 16;
    if (m0 >= M) return;                 // uniform per wave: EXEC stays all-1s

    const int g = lane >> 4;             // half-wave group (K sub-range)
    const int r = lane & 15;             // row (A) / col (B,C) within tile
    const _Float16* __restrict__ aRow = A + (long)(m0 + r) * K + 16 * g;

#pragma unroll
    for (int t0 = 0; t0 < NT; t0 += GROUP) {
        v8f acc[GROUP];
#pragma unroll
        for (int t = 0; t < GROUP; ++t)
#pragma unroll
            for (int v = 0; v < 8; ++v) acc[t][v] = 0.f;

        const _Float16* __restrict__ bBase =
            Bt + (long)(t0 * 16 + r) * K + 16 * g;

        for (int kb = 0; kb < K; kb += 32) {
            // A fragment (16x32 f16): lane holds row m0+r,
            // K = kb+16g .. kb+16g+15 (contiguous)
            union { v16h v; v8h h[2]; } af;
            af.h[0] = *(const v8h*)(aRow + kb);
            af.h[1] = *(const v8h*)(aRow + kb + 8);
#pragma unroll
            for (int t = 0; t < GROUP; ++t) {
                // B fragment (32x16 f16): lane holds col (t0+t)*16+r,
                // K = kb+16g .. kb+16g+15 (contiguous in Bt)
                union { v16h v; v8h h[2]; } bf;
                const _Float16* bp = bBase + (long)t * 16 * K + kb;
                bf.h[0] = *(const v8h*)(bp);
                bf.h[1] = *(const v8h*)(bp + 8);
                acc[t] = __builtin_amdgcn_wmma_f32_16x16x32_f16(
                    false, af.v, false, bf.v, (short)0, acc[t], false, false);
            }
        }

        // Epilogue: C/D layout -> lane (g,r): VGPR v holds (m0+v+8g, col r)
#pragma unroll
        for (int t = 0; t < GROUP; ++t) {
            const int col = (t0 + t) * 16 + r;
            const float bv = bias ? bias[col] : 0.f;
            float* __restrict__ cp = C + (long)(m0 + 8 * g) * Ncols + col;
#pragma unroll
            for (int v = 0; v < 8; ++v) {
                float val = acc[t][v] + bv;
                if (accum) val += cp[(long)v * Ncols];
                if (act == 1) val = val > 0.f ? val : 0.f;
                else if (act == 2) val = 1.f / (1.f + __expf(-val));
                cp[(long)v * Ncols] = val;
            }
        }
    }
}

// ---------------------------- small utility kernels ------------------------
__global__ void fill_f32_kernel(float* p, float v, int n) {
    int i = blockIdx.x * blockDim.x + threadIdx.x;
    if (i < n) p[i] = v;
}
__global__ void fill_u32_kernel(unsigned* p, unsigned v, int n) {
    int i = blockIdx.x * blockDim.x + threadIdx.x;
    if (i < n) p[i] = v;
}
__global__ void f32_to_f16_kernel(const float* __restrict__ s,
                                  _Float16* __restrict__ d, int n) {
    int i = blockIdx.x * blockDim.x + threadIdx.x;
    if (i < n) d[i] = (_Float16)s[i];
}
// W[K,N] f32 -> Wt[N,K] f16
__global__ void transpose_f16_kernel(const float* __restrict__ W,
                                     _Float16* __restrict__ Wt, int K, int Nc) {
    int i = blockIdx.x * blockDim.x + threadIdx.x;
    if (i >= K * Nc) return;
    int n = i / K, k = i - n * K;
    Wt[(long)n * K + k] = (_Float16)W[(long)k * Nc + n];
}

// ------------------------------- GAT pieces --------------------------------
// per-(node,head) attention logits:  als = <h[n,head,:], a_src[head,:]>
__global__ void compute_al_kernel(const float* __restrict__ h,
                                  const float* __restrict__ as,
                                  const float* __restrict__ ad,
                                  float* __restrict__ als,
                                  float* __restrict__ ald, int total) {
    int i = blockIdx.x * blockDim.x + threadIdx.x;
    if (i >= total) return;                       // total = N*HEADS
    int node = i / HEADS, head = i - node * HEADS;
    const float* hp = h + (long)node * FEAT + head * HID;
    const float* sp = as + head * HID;
    const float* dp = ad + head * HID;
    float s0 = 0.f, s1 = 0.f;
#pragma unroll 8
    for (int c = 0; c < HID; ++c) {
        float hv = hp[c];
        s0 += hv * sp[c];
        s1 += hv * dp[c];
    }
    als[i] = s0;
    ald[i] = s1;
}

__device__ __forceinline__ void edge_src_dst(const long long* ei, int e,
                                             int& src, int& dst) {
    if (e < NEDGES) { src = (int)ei[e]; dst = (int)ei[NEDGES + e]; }
    else            { src = dst = e - NEDGES; }
}

__global__ void edge_max_kernel(const long long* __restrict__ ei,
                                const float* __restrict__ als,
                                const float* __restrict__ ald,
                                unsigned* __restrict__ m) {
    int e = blockIdx.x * blockDim.x + threadIdx.x;
    if (e >= NEDGES_SL) return;
    int src, dst; edge_src_dst(ei, e, src, dst);
#pragma unroll
    for (int hh = 0; hh < HEADS; ++hh) {
        float v = als[src * HEADS + hh] + ald[dst * HEADS + hh];
        v = v > 0.f ? v : NEG_SLOPE * v;
        atomicMax(&m[dst * HEADS + hh], fenc(v));
    }
}

__global__ void edge_expsum_kernel(const long long* __restrict__ ei,
                                   const float* __restrict__ als,
                                   const float* __restrict__ ald,
                                   const unsigned* __restrict__ m,
                                   float* __restrict__ p,
                                   float* __restrict__ z) {
    int e = blockIdx.x * blockDim.x + threadIdx.x;
    if (e >= NEDGES_SL) return;
    int src, dst; edge_src_dst(ei, e, src, dst);
#pragma unroll
    for (int hh = 0; hh < HEADS; ++hh) {
        float v = als[src * HEADS + hh] + ald[dst * HEADS + hh];
        v = v > 0.f ? v : NEG_SLOPE * v;
        float pv = __expf(v - fdec(m[dst * HEADS + hh]));
        p[(long)e * HEADS + hh] = pv;
        atomicAdd(&z[dst * HEADS + hh], pv);
    }
}

// one 192-thread block per edge: coalesced gather of h[src,:], scatter-add
__global__ void edge_scatter_kernel(const long long* __restrict__ ei,
                                    const float* __restrict__ h,
                                    const float* __restrict__ p,
                                    const float* __restrict__ z,
                                    float* __restrict__ acc) {
    int e = blockIdx.x;
    int t = threadIdx.x;                      // 0..191
    int src, dst; edge_src_dst(ei, e, src, dst);
    int hh = t >> 6;                          // head = feature/64
    float alpha = p[(long)e * HEADS + hh] / (z[dst * HEADS + hh] + 1e-16f);
    atomicAdd(&acc[(long)dst * FEAT + t], h[(long)src * FEAT + t] * alpha);
}

__global__ void bias_relu_kernel(const float* __restrict__ acc,
                                 const float* __restrict__ b,
                                 float* __restrict__ out, int total) {
    int i = blockIdx.x * blockDim.x + threadIdx.x;
    if (i >= total) return;
    float v = acc[i] + b[i % FEAT];
    out[i] = v > 0.f ? v : 0.f;
}

// ------------------------------- SAGE pieces -------------------------------
__global__ void sage_scatter_kernel(const long long* __restrict__ ei,
                                    const float* __restrict__ h,
                                    float* __restrict__ agg,
                                    float* __restrict__ deg) {
    int e = blockIdx.x;                       // over NEDGES (no self loops)
    int t = threadIdx.x;
    int src = (int)ei[e], dst = (int)ei[NEDGES + e];
    atomicAdd(&agg[(long)dst * FEAT + t], h[(long)src * FEAT + t]);
    if (t == 0) atomicAdd(&deg[dst], 1.f);
}
__global__ void sage_div_kernel(float* __restrict__ agg,
                                const float* __restrict__ deg, int total) {
    int i = blockIdx.x * blockDim.x + threadIdx.x;
    if (i >= total) return;
    agg[i] = agg[i] / fmaxf(deg[i / FEAT], 1.f);
}

// ---------------------------------------------------------------------------
extern "C" void kernel_launch(void* const* d_in, const int* in_sizes, int n_in,
                              void* d_out, int out_size, void* d_ws,
                              size_t ws_size, hipStream_t stream) {
    const float*     x   = (const float*)d_in[0];
    const long long* ei  = (const long long*)d_in[1];
    const float* W1  = (const float*)d_in[2];
    const float* a1s = (const float*)d_in[3];
    const float* a1d = (const float*)d_in[4];
    const float* b1  = (const float*)d_in[5];
    const float* W2  = (const float*)d_in[6];
    const float* a2s = (const float*)d_in[7];
    const float* a2d = (const float*)d_in[8];
    const float* b2  = (const float*)d_in[9];
    const float* Wl  = (const float*)d_in[10];
    const float* bl  = (const float*)d_in[11];
    const float* Wr  = (const float*)d_in[12];
    const float* M1  = (const float*)d_in[13];
    const float* mb1 = (const float*)d_in[14];
    const float* M2  = (const float*)d_in[15];
    const float* mb2 = (const float*)d_in[16];
    float* out = (float*)d_out;

    // ---- workspace layout (256B aligned slices) ----
    char* base = (char*)d_ws;
    size_t off = 0;
    auto take = [&](size_t bytes) -> char* {
        char* p = base + off;
        off = (off + bytes + 255) & ~(size_t)255;
        return p;
    };
    float*    hpre = (float*)take((size_t)NNODES * FEAT * 4);
    float*    accb = (float*)take((size_t)NNODES * FEAT * 4);
    float*    hout = (float*)take((size_t)NNODES * FEAT * 4);
    _Float16* fA   = (_Float16*)take((size_t)NNODES * FEAT * 2);
    _Float16* fB   = (_Float16*)take((size_t)NNODES * FEAT * 2);
    float*    als  = (float*)take((size_t)NNODES * HEADS * 4);
    float*    ald  = (float*)take((size_t)NNODES * HEADS * 4);
    unsigned* mbuf = (unsigned*)take((size_t)NNODES * HEADS * 4);
    float*    zbuf = (float*)take((size_t)NNODES * HEADS * 4);
    float*    pbuf = (float*)take((size_t)NEDGES_SL * HEADS * 4);
    float*    deg  = (float*)take((size_t)NNODES * 4);
    float*    T1   = (float*)take((size_t)NNODES * HID * 4);
    _Float16* W1t  = (_Float16*)take((size_t)INC * FEAT * 2);
    _Float16* W2t  = (_Float16*)take((size_t)FEAT * FEAT * 2);
    _Float16* Wlt  = (_Float16*)take((size_t)FEAT * FEAT * 2);
    _Float16* Wrt  = (_Float16*)take((size_t)FEAT * FEAT * 2);
    _Float16* M1t  = (_Float16*)take((size_t)FEAT * HID * 2);
    _Float16* M2t  = (_Float16*)take((size_t)HID * OUTC * 2);

    const int TB = 256;
    const int gemmBlocks = cdiv(NNODES / 16, 8);   // 8 waves per 256-thr block

    // ---- prep: f16 copies of x and transposed weights ----
    f32_to_f16_kernel<<<cdiv(NNODES * INC, TB), TB, 0, stream>>>(x, fA, NNODES * INC);
    transpose_f16_kernel<<<cdiv(INC * FEAT, TB), TB, 0, stream>>>(W1, W1t, INC, FEAT);
    transpose_f16_kernel<<<cdiv(FEAT * FEAT, TB), TB, 0, stream>>>(W2, W2t, FEAT, FEAT);
    transpose_f16_kernel<<<cdiv(FEAT * FEAT, TB), TB, 0, stream>>>(Wl, Wlt, FEAT, FEAT);
    transpose_f16_kernel<<<cdiv(FEAT * FEAT, TB), TB, 0, stream>>>(Wr, Wrt, FEAT, FEAT);
    transpose_f16_kernel<<<cdiv(FEAT * HID, TB), TB, 0, stream>>>(M1, M1t, FEAT, HID);
    transpose_f16_kernel<<<cdiv(HID * OUTC, TB), TB, 0, stream>>>(M2, M2t, HID, OUTC);

    // ---- one GAT attention+aggregation pass (hpre -> hout) ----
    auto gat_pass = [&](const float* aS, const float* aD, const float* bias) {
        compute_al_kernel<<<cdiv(NNODES * HEADS, TB), TB, 0, stream>>>(
            hpre, aS, aD, als, ald, NNODES * HEADS);
        fill_u32_kernel<<<cdiv(NNODES * HEADS, TB), TB, 0, stream>>>(
            mbuf, FENC_NEG_INF, NNODES * HEADS);
        fill_f32_kernel<<<cdiv(NNODES * HEADS, TB), TB, 0, stream>>>(
            zbuf, 0.f, NNODES * HEADS);
        fill_f32_kernel<<<cdiv(NNODES * FEAT, TB), TB, 0, stream>>>(
            accb, 0.f, NNODES * FEAT);
        edge_max_kernel<<<cdiv(NEDGES_SL, TB), TB, 0, stream>>>(ei, als, ald, mbuf);
        edge_expsum_kernel<<<cdiv(NEDGES_SL, TB), TB, 0, stream>>>(
            ei, als, ald, mbuf, pbuf, zbuf);
        edge_scatter_kernel<<<NEDGES_SL, FEAT, 0, stream>>>(ei, hpre, pbuf, zbuf, accb);
        bias_relu_kernel<<<cdiv(NNODES * FEAT, TB), TB, 0, stream>>>(
            accb, bias, hout, NNODES * FEAT);
    };

    // ---- GAT layer 1: hpre = x_f16 @ W1 ----
    gemm_wmma_kernel<12><<<gemmBlocks, TB, 0, stream>>>(
        fA, W1t, hpre, nullptr, NNODES, INC, FEAT, 0, 0);
    gat_pass(a1s, a1d, b1);

    // ---- GAT layer 2: hpre = h1_f16 @ W2 ----
    f32_to_f16_kernel<<<cdiv(NNODES * FEAT, TB), TB, 0, stream>>>(hout, fA, NNODES * FEAT);
    gemm_wmma_kernel<12><<<gemmBlocks, TB, 0, stream>>>(
        fA, W2t, hpre, nullptr, NNODES, FEAT, FEAT, 0, 0);
    gat_pass(a2s, a2d, b2);           // hout now = h2

    // ---- SAGE: mean-aggregate neighbors, then h = agg@Wl + bl + h2@Wr ----
    fill_f32_kernel<<<cdiv(NNODES * FEAT, TB), TB, 0, stream>>>(accb, 0.f, NNODES * FEAT);
    fill_f32_kernel<<<cdiv(NNODES, TB), TB, 0, stream>>>(deg, 0.f, NNODES);
    sage_scatter_kernel<<<NEDGES, FEAT, 0, stream>>>(ei, hout, accb, deg);
    sage_div_kernel<<<cdiv(NNODES * FEAT, TB), TB, 0, stream>>>(accb, deg, NNODES * FEAT);
    f32_to_f16_kernel<<<cdiv(NNODES * FEAT, TB), TB, 0, stream>>>(hout, fA, NNODES * FEAT);
    f32_to_f16_kernel<<<cdiv(NNODES * FEAT, TB), TB, 0, stream>>>(accb, fB, NNODES * FEAT);
    gemm_wmma_kernel<12><<<gemmBlocks, TB, 0, stream>>>(
        fA, Wrt, hpre, nullptr, NNODES, FEAT, FEAT, 0, 0);          // h2 @ Wr
    gemm_wmma_kernel<12><<<gemmBlocks, TB, 0, stream>>>(
        fB, Wlt, hpre, bl, NNODES, FEAT, FEAT, 1, 0);               // += agg@Wl + bl

    // ---- MLP head: (h @ M1 + mb1) @ M2 + mb2, sigmoid ----
    f32_to_f16_kernel<<<cdiv(NNODES * FEAT, TB), TB, 0, stream>>>(hpre, fA, NNODES * FEAT);
    gemm_wmma_kernel<4><<<gemmBlocks, TB, 0, stream>>>(
        fA, M1t, T1, mb1, NNODES, FEAT, HID, 0, 0);
    f32_to_f16_kernel<<<cdiv(NNODES * HID, TB), TB, 0, stream>>>(T1, fB, NNODES * HID);
    gemm_wmma_kernel<1><<<gemmBlocks, TB, 0, stream>>>(
        fB, M2t, out, mb2, NNODES, HID, OUTC, 0, 2);

    (void)in_sizes; (void)n_in; (void)out_size; (void)ws_size;
}
